// TemporalFusion_88648124989835
// MI455X (gfx1250) — compile-verified, matrix operand order checked by reading, not run on hardware
//
#include <hip/hip_runtime.h>
#include <hip/hip_bf16.h>
#include <math.h>

#define SEQ     2048
#define BATCH   4
#define MTOK    (SEQ*BATCH)   // 8192 tokens
#define IN_DIM  23
#define HID     64
#define LSTMD   128
#define DMODEL  128
#define DINNER  256
#define DSTATE  64
#define DCONV   4
#define DTRANK  8
#define NLAYERS 4

typedef __attribute__((ext_vector_type(16))) _Float16 v16h;
typedef __attribute__((ext_vector_type(8)))  _Float16 v8h;
typedef __attribute__((ext_vector_type(8)))  float    v8f;

__device__ __forceinline__ float sigf(float x) { return 1.f / (1.f + __expf(-x)); }

// Fragment load: two contiguous b128 loads per lane (ISA 16-bit A/B 16x32 layout).
__device__ __forceinline__ v16h ldfrag(const _Float16* __restrict__ p) {
    const v8h lo = *(const v8h*)p;
    const v8h hi = *(const v8h*)(p + 16);
    return __builtin_shufflevector(lo, hi, 0,1,2,3,4,5,6,7,8,9,10,11,12,13,14,15);
}

// ---------------------------------------------------------------------------
// Pack f32 [M, lda] (first K cols) -> zero-padded f16 [M, Kpad]
// ---------------------------------------------------------------------------
__global__ void pack_a_f16(const float* __restrict__ A, int lda, int K, int Kpad,
                           int M, _Float16* __restrict__ out)
{
    int idx = blockIdx.x * blockDim.x + threadIdx.x;
    if (idx >= M * Kpad) return;
    const int m = idx / Kpad, k = idx - m * Kpad;
    out[idx] = (k < K) ? (_Float16)A[(size_t)m * lda + k] : (_Float16)0.f;
}

// Pack f32 W [N,K] -> zero-padded f16 [Npad, Kpad]   (Npad mult of 64)
__global__ void pack_w_f16(const float* __restrict__ W, int K, int Kpad,
                           int N, int Npad, _Float16* __restrict__ out)
{
    int idx = blockIdx.x * blockDim.x + threadIdx.x;
    if (idx >= Npad * Kpad) return;
    const int n = idx / Kpad, k = idx - n * Kpad;
    out[idx] = (n < N && k < K) ? (_Float16)W[(size_t)n * K + k] : (_Float16)0.f;
}

// ---------------------------------------------------------------------------
// WMMA GEMM on packed f16: C[M,N] = Ap[M,Kpad] @ Wp[Npad,Kpad]^T (+biases)(act)
// One wave computes a 16x64 strip (1 A fragment -> 4 accumulators).
// NK = Kpad/32 is a compile-time constant: the k-loop fully unrolls, so each
// stage gets fresh registers (no copy rotation) and the scheduler hoists the
// next stage's b128 loads under the current WMMAs with partial loadcnt waits.
// ---------------------------------------------------------------------------
template<int NK>
__global__ __launch_bounds__(256) void wmma_gemm_f16(
    const _Float16* __restrict__ Ap,   // [M, NK*32]
    const _Float16* __restrict__ Wp,   // [Npad, NK*32]
    const float* __restrict__ bias1,
    const float* __restrict__ bias2,
    float* __restrict__ C, int ldc,
    int M, int N, int Npad, int act)
{
    const int Kpad   = NK * 32;
    const int lane   = threadIdx.x & 31;
    const int wave   = (blockIdx.x * blockDim.x + threadIdx.x) >> 5;
    const int tilesM = M >> 4;           // M is a multiple of 16 (8192)
    const int strips = Npad >> 6;        // 64-wide N strips
    if (wave >= tilesM * strips) return; // wave-uniform exit: EXEC all-1 for WMMA
    const int mt = wave % tilesM;
    const int ns = wave / tilesM;

    const int lrow = lane & 15;          // A row / W row within tile
    const int kb   = (lane >> 4) * 8;    // per-lane K sub-offset (ISA A/B layout)

    const _Float16* arow = Ap + (size_t)(mt * 16 + lrow) * Kpad + kb;
    const _Float16* wrow = Wp + (size_t)(ns * 64 + lrow) * Kpad + kb;
    const size_t wstep = (size_t)16 * Kpad;   // next 16-col tile of W

    v8f acc0 = {}, acc1 = {}, acc2 = {}, acc3 = {};
#pragma unroll
    for (int s = 0; s < NK; ++s) {
        const int k0 = s * 32;
        const v16h a  = ldfrag(arow + k0);
        const v16h b0 = ldfrag(wrow + k0);
        const v16h b1 = ldfrag(wrow + k0 + wstep);
        const v16h b2 = ldfrag(wrow + k0 + 2 * wstep);
        const v16h b3 = ldfrag(wrow + k0 + 3 * wstep);
        acc0 = __builtin_amdgcn_wmma_f32_16x16x32_f16(false, a, false, b0, (short)0, acc0, false, false);
        acc1 = __builtin_amdgcn_wmma_f32_16x16x32_f16(false, a, false, b1, (short)0, acc1, false, false);
        acc2 = __builtin_amdgcn_wmma_f32_16x16x32_f16(false, a, false, b2, (short)0, acc2, false, false);
        acc3 = __builtin_amdgcn_wmma_f32_16x16x32_f16(false, a, false, b3, (short)0, acc3, false, false);
    }

    v8f accs[4] = { acc0, acc1, acc2, acc3 };
#pragma unroll
    for (int j = 0; j < 4; ++j) {
        const int n = ns * 64 + j * 16 + lrow;   // C/D: lane&15 -> N
        if (n < N) {
            const float bb = (bias1 ? bias1[n] : 0.f) + (bias2 ? bias2[n] : 0.f);
#pragma unroll
            for (int r = 0; r < 8; ++r) {
                const int mm = mt * 16 + r + 8 * (lane >> 4); // vgpr r -> M
                float v = accs[j][r] + bb;
                if (act == 1) v = (v > 20.f) ? v : logf(1.f + __expf(v)); // softplus
                C[(size_t)mm * ldc + n] = v;
            }
        }
    }
}

// ---------------------------------------------------------------------------
// Causal depthwise conv (D_CONV=4) over time + SiLU.  xi = first 256 cols of xz.
// ---------------------------------------------------------------------------
__global__ void conv_silu_kernel(const float* __restrict__ xz,
                                 const float* __restrict__ cw,   // [256,1,4]
                                 const float* __restrict__ cb,   // [256]
                                 float* __restrict__ xc)         // [MTOK,256]
{
    int idx = blockIdx.x * blockDim.x + threadIdx.x;
    if (idx >= MTOK * DINNER) return;
    const int c  = idx & (DINNER - 1);
    const int tb = idx >> 8;
    const int b  = tb & (BATCH - 1);
    const int t  = tb >> 2;
    float s = cb[c];
#pragma unroll
    for (int k = 0; k < DCONV; ++k) {
        const int tt = t - (DCONV - 1) + k;
        if (tt >= 0) s += xz[(size_t)(tt * BATCH + b) * (2 * DINNER) + c] * cw[c * DCONV + k];
    }
    xc[idx] = s * sigf(s);
}

// ---------------------------------------------------------------------------
// Selective scan.  Block = batch element, thread = d_inner channel.
// 64-dim state in VGPRs across all 2048 steps; B/C broadcast via LDS.
// Fused epilogue: y = (ys + xc*D) * silu(z).
// ---------------------------------------------------------------------------
__global__ __launch_bounds__(256) void scan_kernel(
    const float* __restrict__ dbc,   // [MTOK,136]  (dt_r | B | C)
    const float* __restrict__ dtp,   // [MTOK,256]  softplus(dt)
    const float* __restrict__ xc,    // [MTOK,256]
    const float* __restrict__ xz,    // [MTOK,512]  (z = cols 256..511)
    const float* __restrict__ A_log, // [256,64]
    const float* __restrict__ Dskip, // [256]
    float* __restrict__ y)           // [MTOK,256]
{
    const int b = blockIdx.x;
    const int d = threadIdx.x;
    __shared__ float Bs[DSTATE], Cs[DSTATE];

    float Arow[DSTATE], h[DSTATE];
#pragma unroll
    for (int s = 0; s < DSTATE; ++s) { Arow[s] = -__expf(A_log[d * DSTATE + s]); h[s] = 0.f; }
    const float Dd = Dskip[d];

    for (int t = 0; t < SEQ; ++t) {
        const size_t tb = (size_t)(t * BATCH + b);
        __syncthreads();
        if (d < DSTATE) {
            Bs[d] = dbc[tb * 136 + DTRANK + d];
            Cs[d] = dbc[tb * 136 + DTRANK + DSTATE + d];
        }
        __syncthreads();
        const float dt = dtp[tb * DINNER + d];
        const float x  = xc[tb * DINNER + d];
        float acc = 0.f;
#pragma unroll
        for (int s = 0; s < DSTATE; ++s) {
            h[s] = __expf(dt * Arow[s]) * h[s] + dt * Bs[s] * x;
            acc += h[s] * Cs[s];
        }
        const float z = xz[tb * (2 * DINNER) + DINNER + d];
        y[tb * DINNER + d] = (acc + x * Dd) * (z * sigf(z));
    }
}

// ---------------------------------------------------------------------------
// LSTM recurrence, one layer, both directions (blockIdx.x = dir).
// 1024 threads = (gate, batch, unit); 64 Whh weights per thread in VGPRs.
// ---------------------------------------------------------------------------
__global__ __launch_bounds__(1024) void lstm_rec_kernel(
    const float* __restrict__ gates_f,  // [MTOK,256]
    const float* __restrict__ gates_b,  // [MTOK,256]
    const float* __restrict__ Whh,      // [4,2,256,64]
    int layer,
    float* __restrict__ out)            // [MTOK,128]
{
    const int dir = blockIdx.x;
    const int tid = threadIdx.x;
    const int g  = tid >> 8;
    const int bj = tid & 255;
    const int b  = bj >> 6;
    const int j  = bj & 63;
    const float* gates = dir ? gates_b : gates_f;
    const float* Wr = Whh + (((size_t)layer * 2 + dir) * 256 + g * 64 + j) * 64;

    float w[HID];
#pragma unroll
    for (int k = 0; k < HID; ++k) w[k] = Wr[k];

    __shared__ float hS[BATCH * HID];
    __shared__ float gS[4][BATCH * HID];
    if (g == 0) hS[bj] = 0.f;
    float cstate = 0.f;
    __syncthreads();

    for (int step = 0; step < SEQ; ++step) {
        const int t = dir ? (SEQ - 1 - step) : step;
        const size_t tb = (size_t)(t * BATCH + b);
        float acc = gates[tb * 256 + g * 64 + j];
#pragma unroll
        for (int k = 0; k < HID; ++k) acc += hS[b * HID + k] * w[k];
        gS[g][bj] = acc;
        __syncthreads();
        if (g == 0) {
            const float gi = gS[0][bj], gf = gS[1][bj], gg = gS[2][bj], go = gS[3][bj];
            cstate = sigf(gf) * cstate + sigf(gi) * tanhf(gg);
            const float hnew = sigf(go) * tanhf(cstate);
            hS[bj] = hnew;
            out[tb * LSTMD + dir * HID + j] = hnew;
        }
        __syncthreads();
    }
}

// ---------------------------------------------------------------------------
// LayerNorm over concat(lstm,mamba) -> gate scalar -> blend.
// ---------------------------------------------------------------------------
__global__ void gate_kernel(const float* __restrict__ lstm,
                            const float* __restrict__ mamba,
                            const float* __restrict__ lnw, const float* __restrict__ lnb,
                            const float* __restrict__ gW, const float* __restrict__ gb,
                            float* __restrict__ out)
{
    const int tb = blockIdx.x * blockDim.x + threadIdx.x;
    if (tb >= MTOK) return;
    float sum = 0.f, sq = 0.f;
    for (int i = 0; i < 256; ++i) {
        const float v = (i < 128) ? lstm[(size_t)tb * 128 + i] : mamba[(size_t)tb * 128 + i - 128];
        sum += v; sq += v * v;
    }
    const float mu = sum * (1.f / 256.f);
    const float var = sq * (1.f / 256.f) - mu * mu;
    const float rstd = rsqrtf(var + 1e-5f);
    float dot = 0.f;
    for (int i = 0; i < 256; ++i) {
        const float v = (i < 128) ? lstm[(size_t)tb * 128 + i] : mamba[(size_t)tb * 128 + i - 128];
        dot += ((v - mu) * rstd * lnw[i] + lnb[i]) * gW[i];
    }
    const float gate = sigf(dot + gb[0]);
    for (int i = 0; i < 128; ++i)
        out[(size_t)tb * 128 + i] = gate * lstm[(size_t)tb * 128 + i]
                                  + (1.f - gate) * mamba[(size_t)tb * 128 + i];
}

// ---------------------------------------------------------------------------
extern "C" void kernel_launch(void* const* d_in, const int* in_sizes, int n_in,
                              void* d_out, int out_size, void* d_ws, size_t ws_size,
                              hipStream_t stream) {
    (void)in_sizes; (void)n_in; (void)out_size; (void)ws_size;
    const float* x          = (const float*)d_in[0];
    const float* lstm_Wih0  = (const float*)d_in[1];
    const float* lstm_WihR  = (const float*)d_in[2];
    const float* lstm_Whh   = (const float*)d_in[3];
    const float* lstm_bih   = (const float*)d_in[4];
    const float* lstm_bhh   = (const float*)d_in[5];
    const float* mproj_W    = (const float*)d_in[6];
    const float* mproj_b    = (const float*)d_in[7];
    const float* in_proj_W  = (const float*)d_in[8];
    const float* conv_w     = (const float*)d_in[9];
    const float* conv_b     = (const float*)d_in[10];
    const float* x_proj_W   = (const float*)d_in[11];
    const float* dt_W       = (const float*)d_in[12];
    const float* dt_b       = (const float*)d_in[13];
    const float* A_log      = (const float*)d_in[14];
    const float* D_skip     = (const float*)d_in[15];
    const float* out_proj_W = (const float*)d_in[16];
    const float* gate_ln_w  = (const float*)d_in[17];
    const float* gate_ln_b  = (const float*)d_in[18];
    const float* gate_W     = (const float*)d_in[19];
    const float* gate_b     = (const float*)d_in[20];
    float* outp = (float*)d_out;

    float* ws = (float*)d_ws;
    size_t o = 0;
    float* xm     = ws + o; o += (size_t)MTOK * 128;
    float* xz     = ws + o; o += (size_t)MTOK * 512;
    float* xc     = ws + o; o += (size_t)MTOK * 256;
    float* dbc    = ws + o; o += (size_t)MTOK * 136;
    float* dtp    = ws + o; o += (size_t)MTOK * 256;
    float* ybuf   = ws + o; o += (size_t)MTOK * 256;
    float* m_out  = ws + o; o += (size_t)MTOK * 128;
    float* bufA   = ws + o; o += (size_t)MTOK * 128;
    float* bufB   = ws + o; o += (size_t)MTOK * 128;
    float* gatesF = ws + o; o += (size_t)MTOK * 256;
    float* gatesB = ws + o; o += (size_t)MTOK * 256;
    // f16 staging (reused serially across GEMMs)
    _Float16* Apk = (_Float16*)(ws + o); o += (size_t)MTOK * 256 / 2;   // 8192x256 halfs
    _Float16* Wpk = (_Float16*)(ws + o); o += (size_t)512 * 128 / 2;    // max 512x128 halfs

    // pack + GEMM helper: C[M,N] = A[:, :K] @ W[N,K]^T
    auto gemm = [&](const float* A, int lda, const float* W,
                    const float* b1, const float* b2,
                    float* C, int ldc, int M, int N, int K, int act,
                    bool repackA) {
        const int Kpad = (K + 31) & ~31;
        const int Npad = (N + 63) & ~63;
        if (repackA) {
            const int na = M * Kpad;
            pack_a_f16<<<(na + 255) / 256, 256, 0, stream>>>(A, lda, K, Kpad, M, Apk);
        }
        const int nw = Npad * Kpad;
        pack_w_f16<<<(nw + 255) / 256, 256, 0, stream>>>(W, K, Kpad, N, Npad, Wpk);
        const int waves  = (M >> 4) * (Npad >> 6);
        const int blocks = (waves + 7) / 8;
        switch (Kpad >> 5) {
        case 1: wmma_gemm_f16<1><<<blocks, 256, 0, stream>>>(Apk, Wpk, b1, b2, C, ldc, M, N, Npad, act); break;
        case 4: wmma_gemm_f16<4><<<blocks, 256, 0, stream>>>(Apk, Wpk, b1, b2, C, ldc, M, N, Npad, act); break;
        default: wmma_gemm_f16<8><<<blocks, 256, 0, stream>>>(Apk, Wpk, b1, b2, C, ldc, M, N, Npad, act); break;
        }
    };

    // ---- Mamba branch ----
    gemm(x,    IN_DIM, mproj_W,    mproj_b, nullptr, xm,   128, MTOK, 128, IN_DIM, 0, true);
    gemm(xm,   128,    in_proj_W,  nullptr, nullptr, xz,   512, MTOK, 512, 128,    0, true);
    conv_silu_kernel<<<(MTOK * DINNER + 255) / 256, 256, 0, stream>>>(xz, conv_w, conv_b, xc);
    gemm(xc,   256,    x_proj_W,   nullptr, nullptr, dbc,  136, MTOK, 136, 256,    0, true);
    gemm(dbc,  136,    dt_W,       dt_b,    nullptr, dtp,  256, MTOK, 256, DTRANK, 1, true);
    scan_kernel<<<BATCH, 256, 0, stream>>>(dbc, dtp, xc, xz, A_log, D_skip, ybuf);
    gemm(ybuf, 256,    out_proj_W, nullptr, nullptr, m_out, 128, MTOK, 128, 256,   0, true);

    // ---- Bidirectional LSTM stack ----
    const float* cur = x; int curld = IN_DIM, curK = IN_DIM;
    float* bufs[2] = { bufA, bufB };
    for (int l = 0; l < NLAYERS; ++l) {
        const float* Wf = (l == 0) ? lstm_Wih0
                                   : lstm_WihR + (((size_t)(l - 1) * 2 + 0) * 256 * 128);
        const float* Wb = (l == 0) ? lstm_Wih0 + (size_t)256 * IN_DIM
                                   : lstm_WihR + (((size_t)(l - 1) * 2 + 1) * 256 * 128);
        // pack A once per layer, reuse for both directions
        gemm(cur, curld, Wf, lstm_bih + (l * 2 + 0) * 256, lstm_bhh + (l * 2 + 0) * 256,
             gatesF, 256, MTOK, 256, curK, 0, true);
        gemm(cur, curld, Wb, lstm_bih + (l * 2 + 1) * 256, lstm_bhh + (l * 2 + 1) * 256,
             gatesB, 256, MTOK, 256, curK, 0, false);
        float* outb = bufs[l & 1];
        lstm_rec_kernel<<<2, 1024, 0, stream>>>(gatesF, gatesB, lstm_Whh, l, outb);
        cur = outb; curld = LSTMD; curK = LSTMD;
    }

    // ---- LayerNorm gate + blend ----
    gate_kernel<<<(MTOK + 255) / 256, 256, 0, stream>>>(cur, m_out, gate_ln_w, gate_ln_b,
                                                        gate_W, gate_b, outp);
}
